// GatedLSM_22892175687823
// MI455X (gfx1250) — matrix operand order repaired and checked by reading, not compiled, not run on hardware
//
#include <hip/hip_runtime.h>
#include <math.h>

// ---------------------------------------------------------------------------
// GatedLSM on MI455X (gfx1250): bf16 WMMA everywhere, 4-tile register
// blocking (one shared A-fragment -> 4 independent WMMAs) and software-
// pipelined fragment loads (next k-block loaded before current WMMAs issue).
//   Phase 0: fp32 -> bf16 weight/x conversion
//   Phase 1: input projection GEMM  I[4096,2048] = x@Wi.T + bi      (WMMA)
//   Phase 2: sequential LIF scan, persistent 32-wave workgroup,
//            spikes in LDS as the WMMA A-fragment layout            (WMMA)
//   Phase 3: RMSNorm (popcount-based, spikes are binary)
//   Phase 4: H = GELU(normed@W1.T + b1)                             (WMMA)
//   Phase 5: out = H@W2.T + b2                                      (WMMA)
// ---------------------------------------------------------------------------

typedef __attribute__((ext_vector_type(16))) __bf16 v16bf;
typedef __attribute__((ext_vector_type(8)))  __bf16 v8bf;
typedef __attribute__((ext_vector_type(8)))  float  v8f;

#define DIMX 1024
#define RES  2048
#define BAT  8
#define TSTEPS 512
#define MROWS (BAT * TSTEPS)      // 4096
#define BETA_C 0.95f
#define SFA_INC_C 0.05f
#define SFA_DECAY_C 0.951229424500714f   // exp(-1/20)
#define TH0_C 1.0f
#define EPS_C 1e-6f

static __device__ __forceinline__ unsigned short f2bf(float f) {
    union { float f; unsigned u; } v; v.f = f;
    unsigned u = v.u;
    unsigned r = u + 0x7FFFu + ((u >> 16) & 1u);   // round-to-nearest-even
    return (unsigned short)(r >> 16);
}

static __device__ __forceinline__ float gelu_exact(float x) {
    return 0.5f * x * (1.0f + erff(x * 0.70710678118654752f));
}

// A-fragment (16x32 bf16) loader: lane l -> row (l&15),
// 16 contiguous K at h*16, h = l>>4 (matches B-fragment order).
static __device__ __forceinline__ v16bf load_afrag(const unsigned short* __restrict__ rowp,
                                                   int k0, int half) {
    v8bf lo = *(const v8bf*)(rowp + k0 + half * 16);
    v8bf hi = *(const v8bf*)(rowp + k0 + half * 16 + 8);
    return __builtin_shufflevector(lo, hi, 0,1,2,3,4,5,6,7,8,9,10,11,12,13,14,15);
}

// B-fragment (32x16 bf16), same k order as A: lane l -> column (l&15), 16 contiguous K
// at k0 + (l>>4)*16.  Column n of B == row n of W (W is [N,K] row-major).
static __device__ __forceinline__ v16bf load_bfrag(const unsigned short* __restrict__ colp,
                                                   int k0, int half) {
    return *(const v16bf*)(colp + k0 + half * 16);
}

static __device__ __forceinline__ v8f wmma_bf16(v16bf a, v16bf b, v8f c) {
    return __builtin_amdgcn_wmma_f32_16x16x32_bf16(false, a, false, b, (short)0, c,
                                                   false, false);
}

// ---------------------------------------------------------------------------
// fp32 -> bf16 conversion
// ---------------------------------------------------------------------------
__global__ __launch_bounds__(256)
void cvt_f32_bf16(const float* __restrict__ in, unsigned short* __restrict__ out, int n) {
    int i = blockIdx.x * 256 + threadIdx.x;
    if (i < n) out[i] = f2bf(in[i]);
}

// ---------------------------------------------------------------------------
// Wave-per-16x64 WMMA GEMM:  C[M,N] = act( A[M,K] * W[N,K]^T + bias )
// A, W bf16 row-major.  Each wave: one A-fragment per k-block shared by
// 4 N-tiles (4 accumulators), software-pipelined loads.
// ---------------------------------------------------------------------------
template<int ACT_GELU, int OUT_BF16>
__global__ __launch_bounds__(256)
void gemm_wmma_bf16(const unsigned short* __restrict__ A,
                    const unsigned short* __restrict__ W,
                    const float* __restrict__ bias,
                    float* __restrict__ Cf,
                    unsigned short* __restrict__ Cb,
                    int Mdim, int Ndim, int Kdim)
{
    const int lane  = threadIdx.x & 31;
    const int wave  = threadIdx.x >> 5;
    const int half  = lane >> 4;
    const int l15   = lane & 15;
    const int grpN  = Ndim >> 6;                       // groups of 4 N-tiles
    const int grp   = blockIdx.x * 8 + wave;
    if (grp >= (Mdim >> 4) * grpN) return;
    const int mtile = grp / grpN;
    const int ngrp  = grp - mtile * grpN;

    const int arow = mtile * 16 + l15;                 // A row served by this lane
    const unsigned short* __restrict__ Ap = A + (size_t)arow * Kdim;
    const unsigned short* __restrict__ Wp[4];
    int col[4];
#pragma unroll
    for (int j = 0; j < 4; ++j) {
        col[j] = ngrp * 64 + j * 16 + l15;             // B/C column served by this lane
        Wp[j]  = W + (size_t)col[j] * Kdim;
    }

    v8f acc[4] = {{}, {}, {}, {}};
    v16bf a_cur = load_afrag(Ap, 0, half);
    v16bf b_cur[4];
#pragma unroll
    for (int j = 0; j < 4; ++j) b_cur[j] = load_bfrag(Wp[j], 0, half);

    for (int k0 = 0; k0 < Kdim - 32; k0 += 32) {
        v16bf a_nxt = load_afrag(Ap, k0 + 32, half);   // pipeline: next k-block
        v16bf b_nxt[4];
#pragma unroll
        for (int j = 0; j < 4; ++j) b_nxt[j] = load_bfrag(Wp[j], k0 + 32, half);
        if (k0 + 512 < Kdim) __builtin_prefetch(Wp[0] + k0 + 512, 0, 0);
#pragma unroll
        for (int j = 0; j < 4; ++j) acc[j] = wmma_bf16(a_cur, b_cur[j], acc[j]);
        a_cur = a_nxt;
#pragma unroll
        for (int j = 0; j < 4; ++j) b_cur[j] = b_nxt[j];
    }
#pragma unroll
    for (int j = 0; j < 4; ++j) acc[j] = wmma_bf16(a_cur, b_cur[j], acc[j]);

    const int mbase = mtile * 16 + half * 8;           // C rows: VGPR r <-> row mbase+r
#pragma unroll
    for (int j = 0; j < 4; ++j) {
        const float bv = bias ? bias[col[j]] : 0.0f;
#pragma unroll
        for (int r = 0; r < 8; ++r) {
            float v = acc[j][r] + bv;
            if (ACT_GELU) v = gelu_exact(v);
            size_t idx = (size_t)(mbase + r) * Ndim + col[j];
            if (OUT_BF16) Cb[idx] = f2bf(v);
            else          Cf[idx] = v;
        }
    }
}

// ---------------------------------------------------------------------------
// Sequential LIF reservoir scan.  ONE workgroup, 32 waves (1024 threads).
// LDS holds spike A-matrix [16 rows x 2048] bf16 (rows 8..15 zero).
// Each wave owns 4 N-tiles (64 neurons): per k-block ONE shared A-fragment
// feeds 4 independent WMMAs.  mem/ath live in registers across all steps.
// ---------------------------------------------------------------------------
__global__ __launch_bounds__(1024)
void reservoir_scan(const float* __restrict__ mem0,
                    const float* __restrict__ ath0,
                    const unsigned short* __restrict__ Wr_bf,   // [RES,RES] bf16
                    const float* __restrict__ I,                // [MROWS,RES] row = b*T+t
                    unsigned short* __restrict__ spk_out,       // [MROWS,RES] bf16
                    float* __restrict__ mem_out,
                    float* __restrict__ ath_out)
{
    __shared__ unsigned short lds_spk[16 * RES];                // 64 KB
    const int lane = threadIdx.x & 31;
    const int wave = threadIdx.x >> 5;
    const int half = lane >> 4;
    const int l15  = lane & 15;

    for (int i = threadIdx.x; i < 16 * RES; i += 1024) lds_spk[i] = 0;  // act0 = 0
    __syncthreads();

    const unsigned short* __restrict__ Wp[4];
    int ncol[4];
#pragma unroll
    for (int j = 0; j < 4; ++j) {
        ncol[j] = (wave * 4 + j) * 16 + l15;
        Wp[j]   = Wr_bf + (size_t)ncol[j] * RES;
    }

    float mem[4][8], ath[4][8];
#pragma unroll
    for (int j = 0; j < 4; ++j)
#pragma unroll
        for (int r = 0; r < 8; ++r) {
            if (half == 0) { mem[j][r] = mem0[r * RES + ncol[j]];
                             ath[j][r] = ath0[r * RES + ncol[j]]; }
            else           { mem[j][r] = 0.0f; ath[j][r] = 0.0f; }
        }

    const unsigned short* __restrict__ Alds = lds_spk + (size_t)l15 * RES;

    for (int t = 0; t < TSTEPS; ++t) {
        v8f acc[4] = {{}, {}, {}, {}};
        v16bf a_cur = load_afrag(Alds, 0, half);
        v16bf b_cur[4];
#pragma unroll
        for (int j = 0; j < 4; ++j) b_cur[j] = load_bfrag(Wp[j], 0, half);

        for (int k0 = 0; k0 < RES - 32; k0 += 32) {
            v16bf a_nxt = load_afrag(Alds, k0 + 32, half);
            v16bf b_nxt[4];
#pragma unroll
            for (int j = 0; j < 4; ++j) b_nxt[j] = load_bfrag(Wp[j], k0 + 32, half);
#pragma unroll
            for (int j = 0; j < 4; ++j) acc[j] = wmma_bf16(a_cur, b_cur[j], acc[j]);
            a_cur = a_nxt;
#pragma unroll
            for (int j = 0; j < 4; ++j) b_cur[j] = b_nxt[j];
        }
#pragma unroll
        for (int j = 0; j < 4; ++j) acc[j] = wmma_bf16(a_cur, b_cur[j], acc[j]);

        unsigned short sbits[4][8];
        if (half == 0) {                               // valid batch rows m = r (0..7)
#pragma unroll
            for (int j = 0; j < 4; ++j)
#pragma unroll
                for (int r = 0; r < 8; ++r) {
                    float cur = acc[j][r] + I[((size_t)r * TSTEPS + t) * RES + ncol[j]];
                    float m_  = BETA_C * mem[j][r] + cur;
                    float s   = (m_ - (TH0_C + ath[j][r])) > 0.0f ? 1.0f : 0.0f;
                    mem[j][r] = m_ * (1.0f - s);
                    ath[j][r] = SFA_DECAY_C * ath[j][r] + SFA_INC_C * s;
                    sbits[j][r] = (s > 0.5f) ? (unsigned short)0x3F80 : (unsigned short)0;
                }
        }
        __syncthreads();                               // all A reads of this step done
        if (half == 0) {
#pragma unroll
            for (int j = 0; j < 4; ++j)
#pragma unroll
                for (int r = 0; r < 8; ++r) {
                    lds_spk[(size_t)r * RES + ncol[j]] = sbits[j][r];
                    spk_out[((size_t)r * TSTEPS + t) * RES + ncol[j]] = sbits[j][r];
                }
        }
        __syncthreads();                               // spikes visible for t+1
    }

    if (half == 0) {
#pragma unroll
        for (int j = 0; j < 4; ++j)
#pragma unroll
            for (int r = 0; r < 8; ++r) {
                mem_out[r * RES + ncol[j]] = mem[j][r];
                ath_out[r * RES + ncol[j]] = ath[j][r];
            }
    }
}

// ---------------------------------------------------------------------------
// RMSNorm: spikes are binary -> mean(spk^2) = count/RES.
// normed[row,n] = spk * rsqrt(count/RES + eps) * rms_w[n]   (bf16 out)
// ---------------------------------------------------------------------------
__global__ __launch_bounds__(256)
void rmsnorm_rows(const unsigned short* __restrict__ spk,
                  const float* __restrict__ rms_w,
                  unsigned short* __restrict__ normed)
{
    const int row = blockIdx.x;
    const unsigned short* __restrict__ s = spk + (size_t)row * RES;
    float part = 0.0f;
    for (int j = threadIdx.x; j < RES; j += 256) part += (s[j] ? 1.0f : 0.0f);
    __shared__ float red[256];
    red[threadIdx.x] = part;
    __syncthreads();
    for (int st = 128; st > 0; st >>= 1) {
        if (threadIdx.x < st) red[threadIdx.x] += red[threadIdx.x + st];
        __syncthreads();
    }
    const float scale = rsqrtf(red[0] * (1.0f / (float)RES) + EPS_C);
    for (int j = threadIdx.x; j < RES; j += 256) {
        float v = (s[j] ? scale : 0.0f) * rms_w[j];
        normed[(size_t)row * RES + j] = f2bf(v);
    }
}

// ---------------------------------------------------------------------------
// Host-side launch
// ---------------------------------------------------------------------------
extern "C" void kernel_launch(void* const* d_in, const int* in_sizes, int n_in,
                              void* d_out, int out_size, void* d_ws, size_t ws_size,
                              hipStream_t stream)
{
    (void)in_sizes; (void)n_in; (void)out_size; (void)ws_size;
    const float* x     = (const float*)d_in[0];   // [8,512,1024]
    const float* mem0  = (const float*)d_in[1];   // [8,2048]
    const float* ath0  = (const float*)d_in[2];   // [8,2048]
    const float* Wi    = (const float*)d_in[3];   // [2048,1024]
    const float* bi    = (const float*)d_in[4];   // [2048]
    const float* Wr    = (const float*)d_in[5];   // [2048,2048]
    const float* rms_w = (const float*)d_in[6];   // [2048]
    const float* W1    = (const float*)d_in[7];   // [2048,2048]
    const float* b1    = (const float*)d_in[8];   // [2048]
    const float* W2    = (const float*)d_in[9];   // [1024,2048]
    const float* b2    = (const float*)d_in[10];  // [1024]

    float* out_main = (float*)d_out;                         // [8,512,1024] == [4096,1024]
    float* mem_out  = out_main + (size_t)MROWS * DIMX;       // [8,2048]
    float* ath_out  = mem_out + BAT * RES;                   // [8,2048]

    // Workspace carve (all 256-B aligned by construction)
    char* ws = (char*)d_ws;
    size_t off = 0;
    auto carve = [&](size_t bytes) { char* p = ws + off; off += (bytes + 255) & ~(size_t)255; return p; };
    unsigned short* Wi_bf  = (unsigned short*)carve((size_t)RES * DIMX * 2);      // 4 MB
    unsigned short* Wr_bf  = (unsigned short*)carve((size_t)RES * RES * 2);       // 8 MB
    unsigned short* W1_bf  = (unsigned short*)carve((size_t)RES * RES * 2);       // 8 MB
    unsigned short* W2_bf  = (unsigned short*)carve((size_t)DIMX * RES * 2);      // 4 MB
    unsigned short* x_bf   = (unsigned short*)carve((size_t)MROWS * DIMX * 2);    // 8 MB
    float*          I_f32  = (float*)carve((size_t)MROWS * RES * 4);              // 32 MB
    unsigned short* spk_bf = (unsigned short*)carve((size_t)MROWS * RES * 2);     // 16 MB
    unsigned short* nrm_bf = (unsigned short*)carve((size_t)MROWS * RES * 2);     // 16 MB
    unsigned short* H_bf   = (unsigned short*)carve((size_t)MROWS * RES * 2);     // 16 MB

    // Phase 0: conversions to bf16
    {
        int n;
        n = RES * DIMX;    cvt_f32_bf16<<<(n + 255) / 256, 256, 0, stream>>>(Wi, Wi_bf, n);
        n = RES * RES;     cvt_f32_bf16<<<(n + 255) / 256, 256, 0, stream>>>(Wr, Wr_bf, n);
        n = RES * RES;     cvt_f32_bf16<<<(n + 255) / 256, 256, 0, stream>>>(W1, W1_bf, n);
        n = DIMX * RES;    cvt_f32_bf16<<<(n + 255) / 256, 256, 0, stream>>>(W2, W2_bf, n);
        n = MROWS * DIMX;  cvt_f32_bf16<<<(n + 255) / 256, 256, 0, stream>>>(x, x_bf, n);
    }

    // Phase 1: I = x @ Wi.T + bi   (M=4096, N=2048, K=1024)
    {
        int groups = (MROWS / 16) * (RES / 64);              // 8192
        gemm_wmma_bf16<0, 0><<<groups / 8, 256, 0, stream>>>(
            x_bf, Wi_bf, bi, I_f32, nullptr, MROWS, RES, DIMX);
    }

    // Phase 2: sequential reservoir scan (single persistent workgroup)
    reservoir_scan<<<1, 1024, 0, stream>>>(mem0, ath0, Wr_bf, I_f32,
                                           spk_bf, mem_out, ath_out);

    // Phase 3: RMSNorm rows
    rmsnorm_rows<<<MROWS, 256, 0, stream>>>(spk_bf, rms_w, nrm_bf);

    // Phase 4: H = GELU(normed @ W1.T + b1)   (M=4096, N=2048, K=2048) -> bf16
    {
        int groups = (MROWS / 16) * (RES / 64);              // 8192
        gemm_wmma_bf16<1, 1><<<groups / 8, 256, 0, stream>>>(
            nrm_bf, W1_bf, b1, nullptr, H_bf, MROWS, RES, RES);
    }

    // Phase 5: out = H @ W2.T + b2   (M=4096, N=1024, K=2048) -> fp32
    {
        int groups = (MROWS / 16) * (DIMX / 64);             // 4096
        gemm_wmma_bf16<0, 0><<<groups / 8, 256, 0, stream>>>(
            H_bf, W2_bf, b2, out_main, nullptr, MROWS, DIMX, RES);
    }
}
